// GAT_1322849928006
// MI455X (gfx1250) — matrix operand order repaired and checked by reading, not compiled, not run on hardware
//
#include <hip/hip_runtime.h>
#include <hip/hip_bf16.h>
#include <math.h>

// ---------------------------------------------------------------------------
// GATv2 3-layer network for MI455X (gfx1250, wave32, WMMA).
//  - Node transforms: f16 WMMA GEMMs (v_wmma_f32_16x16x32_f16), f32 accum.
//    128x64 block tile, 8 waves x 4 accumulators.  Weights converted to f16
//    *column-major* once per layer, so panel staging is 1 global_load_b128 +
//    1 ds_store_b128 per thread per k-step; B fragments are ds_load_b128.
//  - Edge softmax/aggregation: L2-resident gathers + f32 atomics
//    (both 51 MB gather arrays fit in the 192 MB L2 simultaneously).
// ---------------------------------------------------------------------------

typedef __attribute__((ext_vector_type(16))) _Float16 v16h;
typedef __attribute__((ext_vector_type(8)))  _Float16 v8h;
typedef __attribute__((ext_vector_type(8)))  float    v8f;

#define NEG_SLOPE 0.2f

// ----- order-preserving float <-> uint encoding for atomicMax-based segmax --
__device__ __forceinline__ unsigned f2ord(float f) {
    unsigned u = __float_as_uint(f);
    return (u & 0x80000000u) ? ~u : (u | 0x80000000u);
}
__device__ __forceinline__ float ord2f(unsigned u) {
    unsigned b = (u & 0x80000000u) ? (u & 0x7fffffffu) : ~u;
    return __uint_as_float(b);
}
#define ORD_NEG_INF 0x007FFFFFu   // f2ord(-inf)

// ---------------------------------------------------------------------------
// utility kernels
// ---------------------------------------------------------------------------
__global__ void k_f32_to_f16(const float* __restrict__ in,
                             _Float16* __restrict__ out, size_t n) {
    size_t i = (size_t)blockIdx.x * blockDim.x + threadIdx.x;
    size_t stride = (size_t)gridDim.x * blockDim.x;
    for (; i < n; i += stride) out[i] = (_Float16)in[i];
}

// W [K,256] f32 row-major  ->  Wt [256,K] f16 column-major (n-major)
__global__ void k_w_to_f16_t(const float* __restrict__ in,
                             _Float16* __restrict__ out, int K) {
    int i = blockIdx.x * blockDim.x + threadIdx.x;
    if (i >= K * 256) return;
    int k = i >> 8, n = i & 255;
    out[(size_t)n * K + k] = (_Float16)in[i];
}

__global__ void k_fill_u32(unsigned* __restrict__ p, unsigned v, size_t n) {
    size_t i = (size_t)blockIdx.x * blockDim.x + threadIdx.x;
    size_t stride = (size_t)gridDim.x * blockDim.x;
    for (; i < n; i += stride) p[i] = v;
}

// ---------------------------------------------------------------------------
// WMMA GEMM:  Y[N,256] = X[N,K] @ W[K,256] + bias
//   X: f16 row-major.  Wt: f16 column-major [256,K].  Y, bias: f32.
// block = 256 threads = 8 waves.  Block tile: 128 rows x 64 cols.
//   wave w -> rows [blockIdx.x*128 + w*16, +16), all 64 cols (4 accumulators)
//   blockIdx.y in [0,4) -> 64-wide column panel.
// K in {128, 256}, multiple of 32.
// ---------------------------------------------------------------------------
__global__ void k_gemm_wmma(const _Float16* __restrict__ X,
                            const _Float16* __restrict__ Wt,
                            const float* __restrict__ bias,
                            float* __restrict__ Y, int N, int K) {
    // n-major panel: sB[n*32 + k], n in [0,64), k in [0,32)  (4 KB)
    __shared__ _Float16 sB[64 * 32];
    const int wave  = threadIdx.x >> 5;
    const int lane  = threadIdx.x & 31;
    const int mBase = blockIdx.x * 128 + wave * 16;
    const int nBase = blockIdx.y * 64;
    const int lh    = lane >> 4;               // lane-half 0/1
    const int ll    = lane & 15;

    // staging coordinates (constant across k-steps)
    const int sn = threadIdx.x & 63;           // panel column this thread stages
    const int sk = (threadIdx.x >> 6) * 8;     // 8-deep k chunk
    const _Float16* wcol = Wt + (size_t)(nBase + sn) * K + sk;

    v8f acc0 = {}, acc1 = {}, acc2 = {}, acc3 = {};

    for (int kb = 0; kb < K; kb += 32) {
        // ---- stage 32x64 W panel: contiguous 16B load + contiguous 16B store
        *(v8h*)(sB + sn * 32 + sk) = *(const v8h*)(wcol + kb);
        if (kb + 32 < K)                       // each lane prefetches its own
            __builtin_prefetch(wcol + kb + 32, 0, 1);   // next column chunk
        __syncthreads();

        // ---- A fragment: row = mBase + ll; lane-half picks K chunks
        // {0..15} / {16..31}.  Two aligned 16B vector loads.
        int row = mBase + ll;
        if (row >= N) row = N - 1;             // clamp (stores are guarded)
        const _Float16* arow = X + (size_t)row * K + kb + lh * 16;
        v8h alo = *(const v8h*)arow;
        v8h ahi = *(const v8h*)(arow + 8);
        v16h a = __builtin_shufflevector(alo, ahi,
                     0, 1, 2, 3, 4, 5, 6, 7, 8, 9, 10, 11, 12, 13, 14, 15);

        // ---- 4 B fragments (each: 32B contiguous LDS -> 2x ds_load_b128)
        // B frag layout: lane holds n = t*16+ll, k = lh*16 + e (contiguous).
        v16h b0 = *(const v16h*)(sB + (0 * 16 + ll) * 32 + lh * 16);
        v16h b1 = *(const v16h*)(sB + (1 * 16 + ll) * 32 + lh * 16);
        v16h b2 = *(const v16h*)(sB + (2 * 16 + ll) * 32 + lh * 16);
        v16h b3 = *(const v16h*)(sB + (3 * 16 + ll) * 32 + lh * 16);

        acc0 = __builtin_amdgcn_wmma_f32_16x16x32_f16(false, a, false, b0, (short)0, acc0, false, false);
        acc1 = __builtin_amdgcn_wmma_f32_16x16x32_f16(false, a, false, b1, (short)0, acc1, false, false);
        acc2 = __builtin_amdgcn_wmma_f32_16x16x32_f16(false, a, false, b2, (short)0, acc2, false, false);
        acc3 = __builtin_amdgcn_wmma_f32_16x16x32_f16(false, a, false, b3, (short)0, acc3, false, false);
        __syncthreads();
    }

    // ---- store: C/D layout: VGPR r -> M = r + 8*lane-half, N = lane&15
    #pragma unroll
    for (int r = 0; r < 8; ++r) {
        int m = mBase + r + 8 * lh;
        if (m < N) {
            float* yrow = Y + (size_t)m * 256 + nBase;
            yrow[ 0 + ll] = acc0[r] + bias[nBase +  0 + ll];
            yrow[16 + ll] = acc1[r] + bias[nBase + 16 + ll];
            yrow[32 + ll] = acc2[r] + bias[nBase + 32 + ll];
            yrow[48 + ll] = acc3[r] + bias[nBase + 48 + ll];
        }
    }
}

// ---------------------------------------------------------------------------
// edge kernels for H=8, C=32 layers.  One wave per edge, lane = channel.
// Edge e >= E is the self-loop of node (e - E).
// ---------------------------------------------------------------------------
__global__ void k_score(const float* __restrict__ xl, const float* __restrict__ xr,
                        const float* __restrict__ att, const int* __restrict__ ei,
                        int E, int Etot,
                        float* __restrict__ score, unsigned* __restrict__ smaxU) {
    int e = blockIdx.x * 8 + (threadIdx.x >> 5);
    if (e >= Etot) return;
    int lane = threadIdx.x & 31;
    int s = (e < E) ? ei[e]     : (e - E);
    int d = (e < E) ? ei[E + e] : (e - E);
    const float* pl = xl + (size_t)s * 256;
    const float* pr = xr + (size_t)d * 256;
    #pragma unroll
    for (int h = 0; h < 8; ++h) {
        float v = pl[h * 32 + lane] + pr[h * 32 + lane];
        v = (v > 0.f) ? v : NEG_SLOPE * v;            // LeakyReLU
        v *= att[h * 32 + lane];
        #pragma unroll
        for (int off = 16; off > 0; off >>= 1) v += __shfl_xor(v, off, 32);
        if (lane == 0) {
            score[(size_t)e * 8 + h] = v;
            atomicMax(&smaxU[(size_t)d * 8 + h], f2ord(v));
        }
    }
}

__global__ void k_expsum(float* __restrict__ score, const unsigned* __restrict__ smaxU,
                         float* __restrict__ denom, const int* __restrict__ ei,
                         int E, int Etot) {
    size_t i = (size_t)blockIdx.x * blockDim.x + threadIdx.x;
    if (i >= (size_t)Etot * 8) return;
    int e = (int)(i >> 3), h = (int)(i & 7);
    int d = (e < E) ? ei[E + e] : (e - E);
    float m = ord2f(smaxU[(size_t)d * 8 + h]);
    float ex = __expf(score[i] - m);
    score[i] = ex;
    atomicAdd(&denom[(size_t)d * 8 + h], ex);
}

__global__ void k_agg(const float* __restrict__ score, const float* __restrict__ denom,
                      const float* __restrict__ xl, const int* __restrict__ ei,
                      int E, int Etot, float* __restrict__ acc) {
    int e = blockIdx.x * 8 + (threadIdx.x >> 5);
    if (e >= Etot) return;
    int lane = threadIdx.x & 31;
    int s = (e < E) ? ei[e]     : (e - E);
    int d = (e < E) ? ei[E + e] : (e - E);
    const float* pl = xl + (size_t)s * 256;
    #pragma unroll
    for (int h = 0; h < 8; ++h) {
        float alpha = score[(size_t)e * 8 + h] / (denom[(size_t)d * 8 + h] + 1e-16f);
        atomicAdd(&acc[(size_t)d * 256 + h * 32 + lane], alpha * pl[h * 32 + lane]);
    }
}

// acc + bias -> ELU -> f16 (input of next layer)
__global__ void k_post_elu(const float* __restrict__ acc, const float* __restrict__ bias,
                           _Float16* __restrict__ hF16, size_t n) {
    size_t i = (size_t)blockIdx.x * blockDim.x + threadIdx.x;
    size_t stride = (size_t)gridDim.x * blockDim.x;
    for (; i < n; i += stride) {
        float v = acc[i] + bias[i & 255];
        v = (v > 0.f) ? v : (__expf(v) - 1.f);        // ELU
        hF16[i] = (_Float16)v;
    }
}

// ---------------------------------------------------------------------------
// layer 3 (H=1, C=2): tiny transform + edge kernels, per-thread
// ---------------------------------------------------------------------------
__global__ void k_gemm2(const _Float16* __restrict__ X, const float* __restrict__ W,
                        const float* __restrict__ bias, float* __restrict__ Y, int N) {
    int n = blockIdx.x * blockDim.x + threadIdx.x;
    if (n >= N) return;
    const _Float16* xr = X + (size_t)n * 256;
    float a0 = bias[0], a1 = bias[1];
    #pragma unroll 8
    for (int k = 0; k < 256; ++k) {
        float xv = (float)xr[k];
        a0 = fmaf(xv, W[k * 2 + 0], a0);
        a1 = fmaf(xv, W[k * 2 + 1], a1);
    }
    Y[(size_t)n * 2 + 0] = a0;
    Y[(size_t)n * 2 + 1] = a1;
}

__global__ void k_score3(const float* __restrict__ xl, const float* __restrict__ xr,
                         const float* __restrict__ att, const int* __restrict__ ei,
                         int E, int Etot, float* __restrict__ score,
                         unsigned* __restrict__ smaxU) {
    int e = blockIdx.x * blockDim.x + threadIdx.x;
    if (e >= Etot) return;
    int s = (e < E) ? ei[e]     : (e - E);
    int d = (e < E) ? ei[E + e] : (e - E);
    float v0 = xl[(size_t)s * 2 + 0] + xr[(size_t)d * 2 + 0];
    float v1 = xl[(size_t)s * 2 + 1] + xr[(size_t)d * 2 + 1];
    v0 = (v0 > 0.f) ? v0 : NEG_SLOPE * v0;
    v1 = (v1 > 0.f) ? v1 : NEG_SLOPE * v1;
    float sc = att[0] * v0 + att[1] * v1;
    score[e] = sc;
    atomicMax(&smaxU[d], f2ord(sc));
}

__global__ void k_expsum3(float* __restrict__ score, const unsigned* __restrict__ smaxU,
                          float* __restrict__ denom, const int* __restrict__ ei,
                          int E, int Etot) {
    int e = blockIdx.x * blockDim.x + threadIdx.x;
    if (e >= Etot) return;
    int d = (e < E) ? ei[E + e] : (e - E);
    float ex = __expf(score[e] - ord2f(smaxU[d]));
    score[e] = ex;
    atomicAdd(&denom[d], ex);
}

__global__ void k_agg3(const float* __restrict__ score, const float* __restrict__ denom,
                       const float* __restrict__ xl, const int* __restrict__ ei,
                       int E, int Etot, float* __restrict__ out2) {
    int e = blockIdx.x * blockDim.x + threadIdx.x;
    if (e >= Etot) return;
    int s = (e < E) ? ei[e]     : (e - E);
    int d = (e < E) ? ei[E + e] : (e - E);
    float alpha = score[e] / (denom[d] + 1e-16f);
    atomicAdd(&out2[(size_t)d * 2 + 0], alpha * xl[(size_t)s * 2 + 0]);
    atomicAdd(&out2[(size_t)d * 2 + 1], alpha * xl[(size_t)s * 2 + 1]);
}

__global__ void k_final(const float* __restrict__ out2, const float* __restrict__ b2,
                        float* __restrict__ dout, int N) {
    int n = blockIdx.x * blockDim.x + threadIdx.x;
    if (n >= N) return;
    float v0 = out2[(size_t)n * 2 + 0] + b2[0];
    float v1 = out2[(size_t)n * 2 + 1] + b2[1];
    float m = fmaxf(v0, v1);
    float l = logf(__expf(v0 - m) + __expf(v1 - m));
    dout[(size_t)n * 2 + 0] = v0 - m - l;
    dout[(size_t)n * 2 + 1] = v1 - m - l;
}

// ---------------------------------------------------------------------------
// host driver
// ---------------------------------------------------------------------------
extern "C" void kernel_launch(void* const* d_in, const int* in_sizes, int n_in,
                              void* d_out, int out_size, void* d_ws, size_t ws_size,
                              hipStream_t stream) {
    const float* x    = (const float*)d_in[0];
    const int*   ei   = (const int*)d_in[1];
    const float* Wl1  = (const float*)d_in[2];  const float* bl1 = (const float*)d_in[3];
    const float* Wr1  = (const float*)d_in[4];  const float* br1 = (const float*)d_in[5];
    const float* att1 = (const float*)d_in[6];  const float* b1  = (const float*)d_in[7];
    const float* Wl3  = (const float*)d_in[8];  const float* bl3 = (const float*)d_in[9];
    const float* Wr3  = (const float*)d_in[10]; const float* br3 = (const float*)d_in[11];
    const float* att3 = (const float*)d_in[12]; const float* b3  = (const float*)d_in[13];
    const float* Wl2  = (const float*)d_in[14]; const float* bl2 = (const float*)d_in[15];
    const float* Wr2  = (const float*)d_in[16]; const float* br2 = (const float*)d_in[17];
    const float* att2 = (const float*)d_in[18]; const float* b2  = (const float*)d_in[19];

    const int N    = in_sizes[0] / 128;
    const int E    = in_sizes[1] / 2;
    const int Etot = E + N;

    // ---- workspace carve-out ----
    char* base = (char*)d_ws;
    size_t o = 0;
    auto carve = [&](size_t bytes) { size_t r = o; o = (o + bytes + 255) & ~(size_t)255; return r; };
    _Float16* hF16  = (_Float16*)(base + carve((size_t)N * 256 * 2));
    _Float16* wl16  = (_Float16*)(base + carve(256 * 256 * 2));   // column-major f16
    _Float16* wr16  = (_Float16*)(base + carve(256 * 256 * 2));   // column-major f16
    float*    xl    = (float*)(base + carve((size_t)N * 256 * 4));
    float*    xr    = (float*)(base + carve((size_t)N * 256 * 4));
    float*    acc   = (float*)(base + carve((size_t)N * 256 * 4));
    float*    score = (float*)(base + carve((size_t)Etot * 8 * 4));
    unsigned* smaxU = (unsigned*)(base + carve((size_t)N * 8 * 4));
    float*    denom = (float*)(base + carve((size_t)N * 8 * 4));
    float*    xl2   = (float*)(base + carve((size_t)N * 2 * 4));
    float*    xr2   = (float*)(base + carve((size_t)N * 2 * 4));
    float*    out2  = (float*)(base + carve((size_t)N * 2 * 4));
    (void)ws_size;

    const int TB = 256;
    dim3 gGemm((N + 127) / 128, 4);                   // 128-row x 64-col tiles
    int gEdgeWave = (Etot + 7) / 8;                   // wave-per-edge kernels
    int gEdgeThr  = (Etot + TB - 1) / TB;             // thread-per-edge kernels
    int gEH       = (int)(((size_t)Etot * 8 + TB - 1) / TB);
    int gElem     = (int)(((size_t)N * 256 + TB - 1) / TB);
    int gNode     = (N + TB - 1) / TB;

    auto run_layer = [&](const float* Wlf, const float* blf, const float* Wrf,
                         const float* brf, const float* att, const float* bias,
                         int K) {
        int gW = (K * 256 + TB - 1) / TB;
        k_w_to_f16_t<<<gW, TB, 0, stream>>>(Wlf, wl16, K);
        k_w_to_f16_t<<<gW, TB, 0, stream>>>(Wrf, wr16, K);
        k_gemm_wmma<<<gGemm, TB, 0, stream>>>(hF16, wl16, blf, xl, N, K);
        k_gemm_wmma<<<gGemm, TB, 0, stream>>>(hF16, wr16, brf, xr, N, K);
        k_fill_u32<<<256, TB, 0, stream>>>(smaxU, ORD_NEG_INF, (size_t)N * 8);
        k_fill_u32<<<256, TB, 0, stream>>>((unsigned*)denom, 0u, (size_t)N * 8);
        k_fill_u32<<<256, TB, 0, stream>>>((unsigned*)acc, 0u, (size_t)N * 256);
        k_score<<<gEdgeWave, TB, 0, stream>>>(xl, xr, att, ei, E, Etot, score, smaxU);
        k_expsum<<<gEH, TB, 0, stream>>>(score, smaxU, denom, ei, E, Etot);
        k_agg<<<gEdgeWave, TB, 0, stream>>>(score, denom, xl, ei, E, Etot, acc);
        k_post_elu<<<gElem, TB, 0, stream>>>(acc, bias, hF16, (size_t)N * 256);
    };

    // layer 1: input x (N x 128) -> f16
    k_f32_to_f16<<<512, TB, 0, stream>>>(x, hF16, (size_t)N * 128);
    run_layer(Wl1, bl1, Wr1, br1, att1, b1, 128);
    // layer 2
    run_layer(Wl3, bl3, Wr3, br3, att3, b3, 256);

    // layer 3: H=1, C=2
    k_gemm2<<<gNode, TB, 0, stream>>>(hF16, Wl2, bl2, xl2, N);
    k_gemm2<<<gNode, TB, 0, stream>>>(hF16, Wr2, br2, xr2, N);
    k_fill_u32<<<64, TB, 0, stream>>>(smaxU, ORD_NEG_INF, (size_t)N);
    k_fill_u32<<<64, TB, 0, stream>>>((unsigned*)denom, 0u, (size_t)N);
    k_fill_u32<<<64, TB, 0, stream>>>((unsigned*)out2, 0u, (size_t)N * 2);
    k_score3<<<gEdgeThr, TB, 0, stream>>>(xl2, xr2, att2, ei, E, Etot, score, smaxU);
    k_expsum3<<<gEdgeThr, TB, 0, stream>>>(score, smaxU, denom, ei, E, Etot);
    k_agg3<<<gEdgeThr, TB, 0, stream>>>(score, denom, xl2, ei, E, Etot, out2);
    k_final<<<gNode, TB, 0, stream>>>(out2, b2, (float*)d_out, N);
}
